// MaskUnitAttention_2190433321139
// MI455X (gfx1250) — matrix operand order repaired and checked by reading, not compiled, not run on hardware
//
#include <hip/hip_runtime.h>

typedef __attribute__((ext_vector_type(16))) _Float16 v16h;
typedef __attribute__((ext_vector_type(8)))  _Float16 v8h;
typedef __attribute__((ext_vector_type(4)))  _Float16 v4h;
typedef __attribute__((ext_vector_type(8)))  float    v8f;
typedef __attribute__((ext_vector_type(4)))  float    v4f;

#define HEADS    12
#define HEAD_DIM 64
#define NWIN     32
#define SEQ      256          // pre-pool tokens per window
#define QWIN     64           // post-pool queries per window
#define BATCH    8
#define NTOK     8192
#define MROWS    (BATCH * NTOK)      // 65536
#define QKV_N    2304
#define DIMK     768
#define ATT_SCALE 0.125f             // 64^-0.5
#define WIN_ELEMS (SEQ * HEAD_DIM)   // 16384 per (b,h,w)

// ---- WMMA fragment loaders (CDNA5 16x16x32 f16 layouts) ----
// A fragment: lane holds row M=lane&15; K = {8*kh+i, 16+8*kh+i}
__device__ inline v16h frag_a(const _Float16* p) {
  v8h lo = *(const v8h*)p;          // K = kh*8 .. kh*8+7
  v8h hi = *(const v8h*)(p + 16);   // K = 16+kh*8 ..
  v16h r;
#pragma unroll
  for (int i = 0; i < 8; ++i) { r[i] = lo[i]; r[i + 8] = hi[i]; }
  return r;
}
// B fragment: lane holds col N=lane&15; K = {8*kh+i, 16+8*kh+i}
__device__ inline v16h frag_b(const _Float16* p) {
  v8h lo = *(const v8h*)p;
  v8h hi = *(const v8h*)(p + 16);
  v16h r;
#pragma unroll
  for (int i = 0; i < 8; ++i) { r[i] = lo[i]; r[i + 8] = hi[i]; }
  return r;
}
__device__ inline v8f wmma16(v16h a, v16h b, v8f c) {
  return __builtin_amdgcn_wmma_f32_16x16x32_f16(false, a, false, b, (short)0, c,
                                                false, false);
}

// Async global->LDS 16B copy (gfx1250, ASYNCcnt-tracked; GV addressing).
__device__ inline void async_copy_b128(unsigned lds_off, const void* gptr) {
  unsigned long long ga = (unsigned long long)(size_t)gptr;
  asm volatile("global_load_async_to_lds_b128 %0, %1, off"
               :: "v"(lds_off), "v"(ga) : "memory");
}
__device__ inline void wait_async0() {
  asm volatile("s_wait_asynccnt 0x0" ::: "memory");
}
__device__ inline unsigned lds_offset(const void* shared_ptr) {
  // Generic LDS pointers carry the LDS byte offset in addr[31:0] (ISA 10.2).
  return (unsigned)(size_t)shared_ptr;
}

// =====================================================================
// Kernel 0: fp32 -> f16 weight pre-conversion (so GEMMs can async-stage B)
// =====================================================================
__global__ void __launch_bounds__(256)
cvt_f16_kernel(const float* __restrict__ src, _Float16* __restrict__ dst, int n8) {
  int i = blockIdx.x * 256 + threadIdx.x;
  if (i < n8) {
    v4f a = *(const v4f*)(src + (size_t)i * 8);
    v4f b = *(const v4f*)(src + (size_t)i * 8 + 4);
    v8h h;
#pragma unroll
    for (int e = 0; e < 4; ++e) { h[e] = (_Float16)a[e]; h[4 + e] = (_Float16)b[e]; }
    *(v8h*)(dst + (size_t)i * 8) = h;
  }
}

// =====================================================================
// Kernel 1: qkv = x @ qkv_w^T + qkv_b, scattered to Qraw/K (s-major) and
//           V transposed (d-major), all f16. B tile async-staged from f16.
// =====================================================================
__global__ void __launch_bounds__(256)
qkv_gemm_kernel(const float* __restrict__ x, const _Float16* __restrict__ wh,
                const float* __restrict__ bias,
                _Float16* __restrict__ Qraw, _Float16* __restrict__ Kbuf,
                _Float16* __restrict__ Vt) {
  __shared__ _Float16 As[128 * 40];   // 128 rows x 32 K, stride 40 (bank pad)
  __shared__ _Float16 Bs[128 * 40];

  const int tid  = threadIdx.x;
  const int lane = tid & 31;
  const int wid  = tid >> 5;
  const int lrow = lane & 15;
  const int kh   = lane >> 4;
  const int m0   = blockIdx.y * 128;
  const int n0   = blockIdx.x * 128;
  const int wm   = wid & 3;    // 4 wave-rows of 32
  const int wn   = wid >> 2;   // 2 wave-cols of 64
  const unsigned bs_lds = lds_offset((const void*)Bs);

  v8f acc[2][4];
#pragma unroll
  for (int i = 0; i < 2; ++i)
#pragma unroll
    for (int j = 0; j < 4; ++j)
#pragma unroll
      for (int e = 0; e < 8; ++e) acc[i][j][e] = 0.0f;

  for (int k0 = 0; k0 < DIMK; k0 += 32) {
    __syncthreads();
    // Async-stage B tile (f16 weights rows n0..n0+127) straight into LDS.
#pragma unroll
    for (int it = 0; it < 2; ++it) {
      int q   = tid + it * 256;           // 512 16B chunks
      int row = q >> 2, c8 = q & 3;
      async_copy_b128(bs_lds + (unsigned)(row * 80 + c8 * 16),
                      wh + (size_t)(n0 + row) * DIMK + k0 + c8 * 8);
    }
    // Stage A tile (x rows m0..m0+127, fp32 -> f16) through VGPRs.
#pragma unroll
    for (int it = 0; it < 4; ++it) {
      int q   = tid + it * 256;           // 1024 float4 chunks
      int row = q >> 3, c4 = q & 7;
      const float* gp = x + (size_t)(m0 + row) * DIMK + k0 + c4 * 4;
      if (k0 + 32 < DIMK) __builtin_prefetch(gp + 32, 0, 0);  // next k-step
      v4f v = *(const v4f*)gp;
      v4h h;
#pragma unroll
      for (int e = 0; e < 4; ++e) h[e] = (_Float16)v[e];
      *(v4h*)(&As[row * 40 + c4 * 4]) = h;
    }
    wait_async0();
    __syncthreads();

    v16h a[2], b[4];
#pragma unroll
    for (int i = 0; i < 2; ++i)
      a[i] = frag_a(&As[(wm * 32 + i * 16 + lrow) * 40 + kh * 8]);
#pragma unroll
    for (int j = 0; j < 4; ++j)
      b[j] = frag_b(&Bs[(wn * 64 + j * 16 + lrow) * 40 + kh * 8]);
#pragma unroll
    for (int i = 0; i < 2; ++i)
#pragma unroll
      for (int j = 0; j < 4; ++j) acc[i][j] = wmma16(a[i], b[j], acc[i][j]);
  }

  // Epilogue: add bias, convert f16, scatter into attention layout.
#pragma unroll
  for (int i = 0; i < 2; ++i) {
#pragma unroll
    for (int j = 0; j < 4; ++j) {
      int c   = n0 + wn * 64 + j * 16 + lrow;   // output channel 0..2303
      int t   = c / DIMK;                       // 0=q 1=k 2=v (tile-uniform)
      int rem = c - t * DIMK;
      int h   = rem >> 6;                       // head (tile-uniform)
      int d   = rem & 63;
      float bv = bias[c];
#pragma unroll
      for (int e = 0; e < 8; ++e) {
        int m  = m0 + wm * 32 + i * 16 + kh * 8 + e;
        int bb = m >> 13;                       // batch
        int nn = m & 8191;
        int s  = nn >> 5;                       // seq-in-window
        int ww = nn & 31;                       // window
        size_t base = (size_t)((bb * HEADS + h) * NWIN + ww) * WIN_ELEMS;
        _Float16 hv = (_Float16)(acc[i][j][e] + bv);
        if (t == 0)      Qraw[base + (size_t)s * HEAD_DIM + d] = hv;
        else if (t == 1) Kbuf[base + (size_t)s * HEAD_DIM + d] = hv;
        else             Vt  [base + (size_t)d * SEQ + s]      = hv;
      }
    }
  }
}

// =====================================================================
// Kernel 2: per (b,h,w): Q strided-max-pool + scale, S = Qp K^T, softmax,
//           O = P V, scattered f16 into proj-GEMM layout.
// =====================================================================
__global__ void __launch_bounds__(128)
attn_kernel(const _Float16* __restrict__ Qraw, const _Float16* __restrict__ Kbuf,
            const _Float16* __restrict__ Vt, _Float16* __restrict__ O) {
  __shared__ _Float16 Qp[QWIN * 72];        // pooled Q, stride 72 halfs
  __shared__ _Float16 P[4 * 16 * 264];      // per-wave P rows, stride 264

  const int tid  = threadIdx.x;
  const int lane = tid & 31;
  const int wid  = tid >> 5;
  const int lrow = lane & 15;
  const int kh   = lane >> 4;
  const int bhw  = blockIdx.x;
  const int ww   = bhw & 31;
  const int h    = (bhw >> 5) % HEADS;
  const int b    = bhw / (HEADS * NWIN);
  const size_t base = (size_t)bhw * WIN_ELEMS;
  const _Float16* Q = Qraw + base;
  const _Float16* K = Kbuf + base;
  const _Float16* V = Vt + base;

  // ---- Q pooling (strided: s in {qq, qq+64, qq+128, qq+192}) + scale ----
  {
    int qq = tid >> 1;
    int dc = (tid & 1) * 32;
#pragma unroll
    for (int part = 0; part < 4; ++part) {
      int dd = dc + part * 8;
      v8h m = *(const v8h*)(Q + (size_t)qq * HEAD_DIM + dd);
#pragma unroll
      for (int j = 1; j < 4; ++j) {
        v8h v = *(const v8h*)(Q + (size_t)(qq + j * QWIN) * HEAD_DIM + dd);
#pragma unroll
        for (int e = 0; e < 8; ++e) m[e] = (v[e] > m[e]) ? v[e] : m[e];
      }
#pragma unroll
      for (int e = 0; e < 8; ++e) m[e] = (_Float16)((float)m[e] * ATT_SCALE);
      *(v8h*)(&Qp[qq * 72 + dd]) = m;
    }
  }
  __syncthreads();

  // ---- S = (Qp*scale) @ K^T : wave handles 16 query rows ----
  v16h aQ[2];
#pragma unroll
  for (int d0 = 0; d0 < 2; ++d0)
    aQ[d0] = frag_a(&Qp[(wid * 16 + lrow) * 72 + d0 * 32 + kh * 8]);

  v8f st[16];
#pragma unroll
  for (int t = 0; t < 16; ++t)
#pragma unroll
    for (int e = 0; e < 8; ++e) st[t][e] = 0.0f;

#pragma unroll
  for (int t = 0; t < 16; ++t) {
#pragma unroll
    for (int d0 = 0; d0 < 2; ++d0) {
      v16h bk = frag_b(K + (size_t)(t * 16 + lrow) * HEAD_DIM + d0 * 32 + kh * 8);
      st[t] = wmma16(aQ[d0], bk, st[t]);
    }
  }

  // ---- softmax over 256 keys (rows split: reg across tiles, lanes in half) ----
  float rmax[8], rsum[8];
#pragma unroll
  for (int e = 0; e < 8; ++e) {
    float m = st[0][e];
#pragma unroll
    for (int t = 1; t < 16; ++t) m = fmaxf(m, st[t][e]);
    rmax[e] = m;
  }
#pragma unroll
  for (int mask = 1; mask < 16; mask <<= 1)
#pragma unroll
    for (int e = 0; e < 8; ++e)
      rmax[e] = fmaxf(rmax[e], __shfl_xor(rmax[e], mask, 32));
#pragma unroll
  for (int e = 0; e < 8; ++e) rsum[e] = 0.0f;
#pragma unroll
  for (int t = 0; t < 16; ++t)
#pragma unroll
    for (int e = 0; e < 8; ++e) {
      float p = __expf(st[t][e] - rmax[e]);
      st[t][e] = p;
      rsum[e] += p;
    }
#pragma unroll
  for (int mask = 1; mask < 16; mask <<= 1)
#pragma unroll
    for (int e = 0; e < 8; ++e) rsum[e] += __shfl_xor(rsum[e], mask, 32);
#pragma unroll
  for (int e = 0; e < 8; ++e) rsum[e] = 1.0f / rsum[e];

  // ---- stage P (f16) to LDS in A-fragment layout ----
  _Float16* Pw = &P[wid * 16 * 264];
#pragma unroll
  for (int t = 0; t < 16; ++t)
#pragma unroll
    for (int e = 0; e < 8; ++e)
      Pw[(kh * 8 + e) * 264 + t * 16 + lrow] = (_Float16)(st[t][e] * rsum[e]);
  __syncthreads();

  // ---- O = P @ V (V stored d-major -> contiguous B fragments) ----
  v8f oacc[4];
#pragma unroll
  for (int j = 0; j < 4; ++j)
#pragma unroll
    for (int e = 0; e < 8; ++e) oacc[j][e] = 0.0f;

#pragma unroll
  for (int ks = 0; ks < 8; ++ks) {
    v16h ap = frag_a(&Pw[lrow * 264 + ks * 32 + kh * 8]);
#pragma unroll
    for (int j = 0; j < 4; ++j) {
      v16h bv = frag_b(V + (size_t)(j * 16 + lrow) * SEQ + ks * 32 + kh * 8);
      oacc[j] = wmma16(ap, bv, oacc[j]);
    }
  }

  // ---- scatter O into [b, q*32+w, h*64+d] f16 ----
#pragma unroll
  for (int j = 0; j < 4; ++j) {
    int d = j * 16 + lrow;
#pragma unroll
    for (int e = 0; e < 8; ++e) {
      int q = wid * 16 + kh * 8 + e;
      size_t row = (size_t)b * (QWIN * NWIN) + q * NWIN + ww;
      O[row * (HEADS * HEAD_DIM) + h * HEAD_DIM + d] = (_Float16)oacc[j][e];
    }
  }
}

// =====================================================================
// Kernel 3: out = O @ proj_w^T + proj_b. Both tiles f16 -> fully async-staged.
// =====================================================================
__global__ void __launch_bounds__(256)
proj_gemm_kernel(const _Float16* __restrict__ A, const _Float16* __restrict__ wh,
                 const float* __restrict__ bias, float* __restrict__ out) {
  __shared__ _Float16 As[128 * 40];
  __shared__ _Float16 Bs[128 * 40];

  const int tid  = threadIdx.x;
  const int lane = tid & 31;
  const int wid  = tid >> 5;
  const int lrow = lane & 15;
  const int kh   = lane >> 4;
  const int m0   = blockIdx.y * 128;
  const int n0   = blockIdx.x * 128;
  const int wm   = wid & 3;
  const int wn   = wid >> 2;
  const unsigned as_lds = lds_offset((const void*)As);
  const unsigned bs_lds = lds_offset((const void*)Bs);

  v8f acc[2][4];
#pragma unroll
  for (int i = 0; i < 2; ++i)
#pragma unroll
    for (int j = 0; j < 4; ++j)
#pragma unroll
      for (int e = 0; e < 8; ++e) acc[i][j][e] = 0.0f;

  for (int k0 = 0; k0 < DIMK; k0 += 32) {
    __syncthreads();
#pragma unroll
    for (int it = 0; it < 2; ++it) {           // A tile: 512 16B chunks
      int q   = tid + it * 256;
      int row = q >> 2, c8 = q & 3;
      async_copy_b128(as_lds + (unsigned)(row * 80 + c8 * 16),
                      A + (size_t)(m0 + row) * DIMK + k0 + c8 * 8);
    }
#pragma unroll
    for (int it = 0; it < 2; ++it) {           // B tile: 512 16B chunks
      int q   = tid + it * 256;
      int row = q >> 2, c8 = q & 3;
      async_copy_b128(bs_lds + (unsigned)(row * 80 + c8 * 16),
                      wh + (size_t)(n0 + row) * DIMK + k0 + c8 * 8);
    }
    wait_async0();
    __syncthreads();

    v16h a[2], b[4];
#pragma unroll
    for (int i = 0; i < 2; ++i)
      a[i] = frag_a(&As[(wm * 32 + i * 16 + lrow) * 40 + kh * 8]);
#pragma unroll
    for (int j = 0; j < 4; ++j)
      b[j] = frag_b(&Bs[(wn * 64 + j * 16 + lrow) * 40 + kh * 8]);
#pragma unroll
    for (int i = 0; i < 2; ++i)
#pragma unroll
      for (int j = 0; j < 4; ++j) acc[i][j] = wmma16(a[i], b[j], acc[i][j]);
  }

#pragma unroll
  for (int i = 0; i < 2; ++i)
#pragma unroll
    for (int j = 0; j < 4; ++j) {
      int n  = n0 + wn * 64 + j * 16 + lrow;
      float bv = bias[n];
#pragma unroll
      for (int e = 0; e < 8; ++e) {
        int m = m0 + wm * 32 + i * 16 + kh * 8 + e;
        out[(size_t)m * DIMK + n] = acc[i][j][e] + bv;
      }
    }
}

// =====================================================================
extern "C" void kernel_launch(void* const* d_in, const int* in_sizes, int n_in,
                              void* d_out, int out_size, void* d_ws, size_t ws_size,
                              hipStream_t stream) {
  (void)in_sizes; (void)n_in; (void)out_size; (void)ws_size;
  const float* x      = (const float*)d_in[0];
  // d_in[1] = attn_mask (unused with mask-unit attention; faithful to reference)
  const float* qkv_w  = (const float*)d_in[2];
  const float* qkv_b  = (const float*)d_in[3];
  const float* proj_w = (const float*)d_in[4];
  const float* proj_b = (const float*)d_in[5];
  float* out = (float*)d_out;

  const size_t per = (size_t)BATCH * HEADS * NWIN * WIN_ELEMS;  // 50,331,648 f16
  _Float16* Qraw  = (_Float16*)d_ws;
  _Float16* Kbuf  = Qraw + per;
  _Float16* Vt    = Kbuf + per;
  _Float16* Of16  = Vt + per;                       // [16384 x 768] f16
  _Float16* Wqkvh = Of16 + (size_t)MROWS / 4 * DIMK / 1;  // placeholder, fixed below
  // Of16 holds 16384*768 halfs:
  Wqkvh = Of16 + (size_t)16384 * DIMK;
  _Float16* Wprjh = Wqkvh + (size_t)QKV_N * DIMK;

  cvt_f16_kernel<<<dim3((QKV_N * DIMK / 8 + 255) / 256), 256, 0, stream>>>(
      qkv_w, Wqkvh, QKV_N * DIMK / 8);
  cvt_f16_kernel<<<dim3((DIMK * DIMK / 8 + 255) / 256), 256, 0, stream>>>(
      proj_w, Wprjh, DIMK * DIMK / 8);
  qkv_gemm_kernel<<<dim3(QKV_N / 128, MROWS / 128), 256, 0, stream>>>(
      x, Wqkvh, qkv_b, Qraw, Kbuf, Vt);
  attn_kernel<<<dim3(BATCH * HEADS * NWIN), 128, 0, stream>>>(
      Qraw, Kbuf, Vt, Of16);
  proj_gemm_kernel<<<dim3(DIMK / 128, 16384 / 128), 256, 0, stream>>>(
      Of16, Wprjh, proj_b, out);
}
